// RGCN_44160853737690
// MI455X (gfx1250) — matrix operand order repaired and checked by reading, not compiled, not run on hardware
//
#include <hip/hip_runtime.h>
#include <hip/hip_bf16.h>

// Problem constants (fixed by the reference)
#define NN     51200
#define EE     819200
#define RR     16
#define NBASE  8
#define HH     128
#define KTOT   1152      // NBASE*HH (bases part) + HH (root part)
#define BQn    512

typedef __attribute__((ext_vector_type(16))) __bf16 v16bf;
typedef __attribute__((ext_vector_type(8)))  float  v8f;

union FragU { uint4 u[2]; v16bf v; };

__device__ __forceinline__ unsigned short f2bf(float f) {
  unsigned int u = __float_as_uint(f);
  u += 0x7FFFu + ((u >> 16) & 1u);     // round-to-nearest-even
  return (unsigned short)(u >> 16);
}

__device__ __forceinline__ unsigned int pack_bf2(float a, float b) {
  return (unsigned int)f2bf(a) | ((unsigned int)f2bf(b) << 16);
}

// One 16x16x32 bf16 WMMA step: A fragment from LDS tile, B fragment from Wt.
// A layout (16-bit 16x32): lane m=L%16, h=L/16; elems 0..7 -> K=8h+e,
//                          elems 8..15 -> K=16+8h+(e-8).
// B layout (16-bit 32x16): lane n=L%16, h=L/16; same K order as A.
__device__ __forceinline__ void wmma_step(v8f& acc,
                                          const unsigned short (*St)[40],
                                          const unsigned short* wk,
                                          int mrow, int hhalf) {
  FragU af, bf;
  af.u[0] = *(const uint4*)&St[mrow][8 * hhalf];
  af.u[1] = *(const uint4*)&St[mrow][16 + 8 * hhalf];
  const unsigned short* wp = wk + 8 * hhalf;
  bf.u[0] = *(const uint4*)(wp);
  bf.u[1] = *(const uint4*)(wp + 16);
  acc = __builtin_amdgcn_wmma_f32_16x16x32_bf16(
      false, af.v, false, bf.v, (short)0, acc, false, false);
}

// Build one 16x32 bf16 S tile (bases part): S[m][j] = sum_r w[m,r]*A[m, r*128+c0+j]
// Each of the 256 threads produces one contiguous channel pair (float2 loads).
__device__ __forceinline__ void build_base_tile(unsigned short (*St)[40],
                                                const float* __restrict__ Ablk,
                                                const float (*comp_s)[NBASE],
                                                const float (*inv_s)[RR],
                                                int kc, int tid) {
  int m  = tid >> 4;
  int j0 = (tid & 15) << 1;
  int b  = kc >> 2;                // base index: k0/128
  int c0 = (kc & 3) << 5;          // channel offset within base
  const float* Arow = Ablk + (size_t)m * (RR * HH) + c0 + j0;
  float sx = 0.0f, sy = 0.0f;
  #pragma unroll
  for (int r = 0; r < RR; ++r) {
    float  w = comp_s[r][b] * inv_s[m][r];
    float2 a = *(const float2*)(Arow + r * HH);
    sx = fmaf(w, a.x, sx);
    sy = fmaf(w, a.y, sy);
  }
  *(unsigned int*)&St[m][j0] = pack_bf2(sx, sy);
}

// Build one 16x32 bf16 S tile (root part): straight copy of hin chunk.
__device__ __forceinline__ void build_root_tile(unsigned short (*St)[40],
                                                const float* __restrict__ hblk,
                                                int kc, int tid) {
  int m  = tid >> 4;
  int j0 = (tid & 15) << 1;
  float2 v = *(const float2*)(hblk + (size_t)m * HH + (kc - 32) * 32 + j0);
  *(unsigned int*)&St[m][j0] = pack_bf2(v.x, v.y);
}

// ---------------------------------------------------------------------------
// h0[n,c] = c < 64 ? x[n,c] : (node_ent[n] == c-64 ? 1 : 0)
__global__ __launch_bounds__(256) void build_h0(const float* __restrict__ x,
                                                const int* __restrict__ ent,
                                                float* __restrict__ h0) {
  int i = blockIdx.x * 256 + threadIdx.x;        // over N*128
  int n = i >> 7, c = i & 127;
  float v;
  if (c < 64) v = x[(size_t)n * 64 + c];
  else        v = (ent[n] == (c - 64)) ? 1.0f : 0.0f;
  h0[(size_t)n * HH + c] = v;
}

// ---------------------------------------------------------------------------
// cnt[dst*R + etype] += 1
__global__ __launch_bounds__(256) void count_kernel(const int* __restrict__ edst,
                                                    const int* __restrict__ etyp,
                                                    int* __restrict__ cnt) {
  int e = blockIdx.x * 256 + threadIdx.x;
  if (e < EE) atomicAdd(&cnt[(size_t)edst[e] * RR + etyp[e]], 1);
}

// ---------------------------------------------------------------------------
// A[(dst*R + etype)*128 + c] += h[src, c]    (2 edges per 256-thread block)
__global__ __launch_bounds__(256) void scatter_agg(const float* __restrict__ h,
                                                   const int* __restrict__ esrc,
                                                   const int* __restrict__ edst,
                                                   const int* __restrict__ etyp,
                                                   float* __restrict__ A) {
  int e = blockIdx.x * 2 + (threadIdx.x >> 7);
  int c = threadIdx.x & 127;
  int s = esrc[e];
  size_t row = (size_t)edst[e] * RR + etyp[e];
  float v = h[(size_t)s * HH + c];
  unsafeAtomicAdd(&A[row * HH + c], v);          // global_atomic_add_f32
}

// ---------------------------------------------------------------------------
// Pack Wcat into bf16, transposed (H-major): Wt[h][k], k<1024 -> bases[b][c][h],
// k>=1024 -> root[k-1024][h]
__global__ __launch_bounds__(256) void prep_w(const float* __restrict__ bases,
                                              const float* __restrict__ root,
                                              unsigned short* __restrict__ Wt) {
  int i = blockIdx.x * 256 + threadIdx.x;        // over KTOT*HH
  int k = i >> 7, hcol = i & 127;
  float v;
  if (k < NBASE * HH) {
    int b = k >> 7, c = k & 127;
    v = bases[((size_t)b * HH + c) * HH + hcol];
  } else {
    v = root[(size_t)(k - NBASE * HH) * HH + hcol];
  }
  Wt[(size_t)hcol * KTOT + k] = f2bf(v);
}

// ---------------------------------------------------------------------------
// Fused combine + GEMM:  hout[16 nodes x 128] = relu( [S|h] @ Wcat + bias )
// S tile built on the fly in LDS (double-buffered), one wave per 16x16 H-tile.
__global__ __launch_bounds__(256) void rgcn_gemm(const float* __restrict__ A,
                                                 const int* __restrict__ cnt,
                                                 const float* __restrict__ comp,
                                                 const float* __restrict__ hin,
                                                 const unsigned short* __restrict__ Wt,
                                                 const float* __restrict__ bias,
                                                 float* __restrict__ hout,
                                                 int relu) {
  __shared__ __align__(16) unsigned short S[2][16][40];  // 2x (16x32 bf16, +pad)
  __shared__ float inv_s[16][RR];
  __shared__ float comp_s[RR][NBASE];

  const int node0 = blockIdx.x * 16;
  const int tid   = threadIdx.x;

  if (tid < RR * NBASE) comp_s[tid >> 3][tid & 7] = comp[tid];
  {
    int m = tid >> 4, r = tid & 15;                      // 256 == 16*16
    int c = cnt[(size_t)(node0 + m) * RR + r];
    inv_s[m][r] = 1.0f / fmaxf((float)c, 1.0f);
  }
  __syncthreads();

  const int lane  = tid & 31;
  const int wave  = tid >> 5;                            // 0..7 -> H cols 16*wave..
  const int ncol  = (wave << 4) + (lane & 15);           // output column h
  const int hhalf = lane >> 4;                           // 0/1
  const int mrow  = lane & 15;

  float bv = bias[ncol];
  v8f acc = {bv, bv, bv, bv, bv, bv, bv, bv};

  const float* Ablk = A + (size_t)node0 * (RR * HH);
  const float* hblk = hin + (size_t)node0 * HH;
  const unsigned short* wrow = Wt + (size_t)ncol * KTOT;

  // ---- phase 1: 8 bases x 128 channels = 32 K-chunks (double-buffered) ----
  for (int kc = 0; kc < 32; kc += 2) {
    build_base_tile(S[0], Ablk, comp_s, inv_s, kc, tid);
    __syncthreads();
    __builtin_prefetch(wrow + (kc + 2) * 32, 0, 1);      // global_prefetch_b8
    wmma_step(acc, S[0], wrow + kc * 32, mrow, hhalf);

    build_base_tile(S[1], Ablk, comp_s, inv_s, kc + 1, tid);
    __syncthreads();
    wmma_step(acc, S[1], wrow + (kc + 1) * 32, mrow, hhalf);
  }
  // ---- phase 2: root block, 4 K-chunks ----
  for (int kc = 32; kc < 36; kc += 2) {
    build_root_tile(S[0], hblk, kc, tid);
    __syncthreads();
    wmma_step(acc, S[0], wrow + kc * 32, mrow, hhalf);

    build_root_tile(S[1], hblk, kc + 1, tid);
    __syncthreads();
    wmma_step(acc, S[1], wrow + (kc + 1) * 32, mrow, hhalf);
  }

  #pragma unroll
  for (int v = 0; v < 8; ++v) {
    int m = v + 8 * hhalf;
    float o = acc[v];
    if (relu) o = fmaxf(o, 0.0f);
    hout[(size_t)(node0 + m) * HH + ncol] = o;
  }
}

// ---------------------------------------------------------------------------
// out[q, c] = [h4[dst[q]+ptr[q]] , rel_table[q_rel[q]]] . lin_w[:,c] + lin_b[c]
__global__ __launch_bounds__(64) void head_kernel(const float* __restrict__ h4,
                                                  const int* __restrict__ qdst,
                                                  const int* __restrict__ qrel,
                                                  const int* __restrict__ ptr,
                                                  const float* __restrict__ rel_table,
                                                  const float* __restrict__ lin_w,
                                                  const float* __restrict__ lin_b,
                                                  float* __restrict__ out) {
  __shared__ float feat[2 * HH];
  int q = blockIdx.x, tid = threadIdx.x;
  int qo = qdst[q] + ptr[q];
  int r  = qrel[q];
  for (int i = tid; i < HH; i += 64) {
    feat[i]      = h4[(size_t)qo * HH + i];
    feat[HH + i] = rel_table[(size_t)r * HH + i];
  }
  __syncthreads();
  if (tid < 2) {
    float s = lin_b[tid];
    for (int k = 0; k < 2 * HH; ++k) s = fmaf(feat[k], lin_w[k * 2 + tid], s);
    out[q * 2 + tid] = s;
  }
}

// ---------------------------------------------------------------------------
extern "C" void kernel_launch(void* const* d_in, const int* in_sizes, int n_in,
                              void* d_out, int out_size, void* d_ws, size_t ws_size,
                              hipStream_t stream) {
  (void)in_sizes; (void)n_in; (void)out_size; (void)ws_size;
  const float* x        = (const float*)d_in[0];
  const int*   node_ent = (const int*)d_in[1];
  const int*   esrc     = (const int*)d_in[2];
  const int*   edst     = (const int*)d_in[3];
  const int*   etyp     = (const int*)d_in[4];
  const int*   qdst     = (const int*)d_in[5];
  const int*   qrel     = (const int*)d_in[6];
  const int*   ptr      = (const int*)d_in[7];
  const float* bases[4] = {(const float*)d_in[8],  (const float*)d_in[12],
                           (const float*)d_in[16], (const float*)d_in[20]};
  const float* comp[4]  = {(const float*)d_in[9],  (const float*)d_in[13],
                           (const float*)d_in[17], (const float*)d_in[21]};
  const float* root[4]  = {(const float*)d_in[10], (const float*)d_in[14],
                           (const float*)d_in[18], (const float*)d_in[22]};
  const float* cbias[4] = {(const float*)d_in[11], (const float*)d_in[15],
                           (const float*)d_in[19], (const float*)d_in[23]};
  const float* rel_table = (const float*)d_in[24];
  const float* lin_w     = (const float*)d_in[25];
  const float* lin_b     = (const float*)d_in[26];
  float* out = (float*)d_out;

  char* ws = (char*)d_ws;
  size_t off = 0;
  auto take = [&](size_t bytes) -> char* {
    char* p = ws + off;
    off = (off + bytes + 255) & ~(size_t)255;
    return p;
  };
  float*          hA  = (float*)take((size_t)NN * HH * 4);
  float*          hB  = (float*)take((size_t)NN * HH * 4);
  int*            cnt = (int*)take((size_t)NN * RR * 4);
  float*          Agg = (float*)take((size_t)NN * RR * HH * 4);
  unsigned short* Wt  = (unsigned short*)take((size_t)HH * KTOT * 2);

  hipMemsetAsync(cnt, 0, (size_t)NN * RR * 4, stream);
  count_kernel<<<EE / 256, 256, 0, stream>>>(edst, etyp, cnt);
  build_h0<<<(NN * HH) / 256, 256, 0, stream>>>(x, node_ent, hA);

  float* hcur = hA;
  float* hnxt = hB;
  for (int l = 0; l < 4; ++l) {
    prep_w<<<(KTOT * HH) / 256, 256, 0, stream>>>(bases[l], root[l], Wt);
    hipMemsetAsync(Agg, 0, (size_t)NN * RR * HH * 4, stream);
    scatter_agg<<<EE / 2, 256, 0, stream>>>(hcur, esrc, edst, etyp, Agg);
    rgcn_gemm<<<NN / 16, 256, 0, stream>>>(Agg, cnt, comp[l], hcur, Wt,
                                           cbias[l], hnxt, (l < 3) ? 1 : 0);
    float* t = hcur; hcur = hnxt; hnxt = t;
  }
  head_kernel<<<BQn, 64, 0, stream>>>(hcur, qdst, qrel, ptr, rel_table,
                                      lin_w, lin_b, out);
}
